// SelfAttention3D_16681652978025
// MI455X (gfx1250) — matrix operand order repaired and checked by reading, not compiled, not run on hardware
//
#include <hip/hip_runtime.h>
#include <hip/hip_bf16.h>

// ---------- types ----------
typedef __bf16 bf16x16 __attribute__((ext_vector_type(16)));
typedef float  f32x8   __attribute__((ext_vector_type(8)));

union Frag {
    bf16x16        v;
    __bf16         e[16];
    unsigned short h[16];
    uint4          q[2];
};

static __device__ __forceinline__ f32x8 wmma_bf16(const Frag& a, const Frag& b, f32x8 c) {
    return __builtin_amdgcn_wmma_f32_16x16x32_bf16(false, a.v, false, b.v,
                                                   (short)0, c, false, false);
}

#define NPOS 4096
#define CCH  256

// ---------- kernel 0: x [256][4096] f32 -> xT [4096][256] bf16 ----------
__global__ void k_xpose(const float* __restrict__ x, __bf16* __restrict__ xT) {
    __shared__ float tile[32][33];
    int p0 = blockIdx.x * 32;            // position tile
    int c0 = blockIdx.y * 32;            // channel tile
    int tx = threadIdx.x;                // 0..31
    int ty = threadIdx.y;                // 0..7
#pragma unroll
    for (int i = 0; i < 4; ++i) {
        int c = ty * 4 + i;
        tile[c][tx] = x[(size_t)(c0 + c) * NPOS + p0 + tx];
    }
    __syncthreads();
#pragma unroll
    for (int i = 0; i < 4; ++i) {
        int p = ty * 4 + i;
        xT[(size_t)(p0 + p) * CCH + c0 + tx] = (__bf16)tile[tx][p];
    }
}

// ---------- kernel 1: QKV projection ----------
// q/k -> transposed bf16 [4096][256] (q has 1/sqrt(d) folded); v -> bf16 [256][4096]
__global__ void k_qkv(const __bf16* __restrict__ xT,
                      const float* __restrict__ Wq, const float* __restrict__ Wk,
                      const float* __restrict__ Wv,
                      const float* __restrict__ bq, const float* __restrict__ bk,
                      const float* __restrict__ bv,
                      __bf16* __restrict__ qT, __bf16* __restrict__ kT,
                      __bf16* __restrict__ vC) {
    const int which = blockIdx.y;
    const float* W    = (which == 0) ? Wq : (which == 1) ? Wk : Wv;
    const float* bias = (which == 0) ? bq : (which == 1) ? bk : bv;
    const int m0 = (blockIdx.x & 15) * 16;       // output-channel tile
    const int n0 = (blockIdx.x >> 4) * 64;       // position tile (4 sub-tiles)
    const int lane  = threadIdx.x & 31;
    const int mrow  = lane & 15;
    const int khalf = lane >> 4;

    f32x8 acc[4];
#pragma unroll
    for (int t = 0; t < 4; ++t)
#pragma unroll
        for (int r = 0; r < 8; ++r) acc[t][r] = 0.0f;

    for (int k0 = 0; k0 < CCH; k0 += 32) {
        Frag A;                                   // 16x32 bf16, A layout
        const float* wr = W + (size_t)(m0 + mrow) * CCH + k0 + khalf * 8;
#pragma unroll
        for (int j = 0; j < 8; ++j) A.e[j]     = (__bf16)wr[j];
#pragma unroll
        for (int j = 0; j < 8; ++j) A.e[8 + j] = (__bf16)wr[16 + j];
#pragma unroll
        for (int t = 0; t < 4; ++t) {
            Frag B;                               // 32x16 bf16, B layout
            const uint4* s = (const uint4*)(xT + (size_t)(n0 + t * 16 + mrow) * CCH
                                               + k0 + khalf * 8);
            B.q[0] = s[0];
            B.q[1] = s[2];
            acc[t] = wmma_bf16(A, B, acc[t]);
        }
    }

    const float scale = (which == 0) ? 0.17677669529663687f : 1.0f;  // 1/sqrt(32)
#pragma unroll
    for (int t = 0; t < 4; ++t) {
        const int pos = n0 + t * 16 + mrow;       // D: lane col = position
        if (which == 2) {                         // v: channel-major
#pragma unroll
            for (int r = 0; r < 8; ++r) {
                int chan = m0 + khalf * 8 + r;
                vC[(size_t)chan * NPOS + pos] = (__bf16)(acc[t][r] + bias[chan]);
            }
        } else {                                  // q/k: position-major (transposed)
            Frag o;
#pragma unroll
            for (int r = 0; r < 8; ++r)
                o.e[r] = (__bf16)((acc[t][r] + bias[m0 + khalf * 8 + r]) * scale);
            __bf16* dst = ((which == 0) ? qT : kT) + (size_t)pos * CCH + m0 + khalf * 8;
            *(uint4*)dst = o.q[0];
        }
    }
}

// ---------- kernel 2: flash attention, one wave per (head, 16-query tile) ----------
__global__ void k_attn(const __bf16* __restrict__ qT,
                       const __bf16* __restrict__ kT,
                       const __bf16* __restrict__ vC,
                       __bf16* __restrict__ attnT) {
    const int head = blockIdx.x >> 8;
    const int q0   = (blockIdx.x & 255) * 16;
    const int lane  = threadIdx.x & 31;
    const int mrow  = lane & 15;
    const int khalf = lane >> 4;
    const int hc    = head * 32;

    // B_q: K=32 (full head dim) x N=16 queries; two runs of 8 (d=8h.., 16+8h..) per lane from qT
    Frag Bq;
    {
        const uint4* s = (const uint4*)(qT + (size_t)(q0 + mrow) * CCH + hc + khalf * 8);
        Bq.q[0] = s[0];
        Bq.q[1] = s[2];
    }

    float mrun = -1e30f, srun = 0.0f;
    f32x8 O0, O1;
#pragma unroll
    for (int r = 0; r < 8; ++r) { O0[r] = 0.0f; O1[r] = 0.0f; }
    f32x8 zero;
#pragma unroll
    for (int r = 0; r < 8; ++r) zero[r] = 0.0f;

    const __bf16* kbase = kT + (size_t)mrow * CCH + hc + khalf * 8;
    const __bf16* vbase0 = vC + (size_t)(hc + mrow) * NPOS + khalf * 8;
    const __bf16* vbase1 = vC + (size_t)(hc + 16 + mrow) * NPOS + khalf * 8;

    for (int k0 = 0; k0 < NPOS; k0 += 32) {
        // prefetch next chunk's K/V lines into cache (global_prefetch_b8)
        if (k0 + 32 < NPOS) {
            __builtin_prefetch(kbase + (size_t)(k0 + 32) * CCH, 0, 3);
            __builtin_prefetch(kbase + (size_t)(k0 + 48) * CCH, 0, 3);
            __builtin_prefetch(vbase0 + k0 + 32, 0, 3);
            __builtin_prefetch(vbase1 + k0 + 32, 0, 3);
        }

        // A_k (two 16-key tiles): A[m][kk] = k[d(kk)][key]; contiguous runs of 8 from kT
        Frag Ak0, Ak1;
        {
            const __bf16* p0 = kbase + (size_t)k0 * CCH;
            Ak0.q[0] = *(const uint4*)p0;
            Ak0.q[1] = *(const uint4*)(p0 + 16);
            const __bf16* p1 = kbase + (size_t)(k0 + 16) * CCH;
            Ak1.q[0] = *(const uint4*)p1;
            Ak1.q[1] = *(const uint4*)(p1 + 16);
        }
        f32x8 S0 = wmma_bf16(Ak0, Bq, zero);      // S^T tile: keys 0-15  x queries
        f32x8 S1 = wmma_bf16(Ak1, Bq, zero);      // S^T tile: keys 16-31 x queries

        // online softmax over this 32-key chunk (lane = one query column)
        float cm = -1e30f;
#pragma unroll
        for (int r = 0; r < 8; ++r) cm = fmaxf(cm, fmaxf(S0[r], S1[r]));
        cm = fmaxf(cm, __shfl_xor(cm, 16, 32));
        const float newm  = fmaxf(mrun, cm);
        const float alpha = __expf(mrun - newm);

        float P0[8], P1[8], csum = 0.0f;
#pragma unroll
        for (int r = 0; r < 8; ++r) {
            P0[r] = __expf(S0[r] - newm);
            P1[r] = __expf(S1[r] - newm);
            csum += P0[r] + P1[r];
        }
        csum += __shfl_xor(csum, 16, 32);
        srun = srun * alpha + csum;
        mrun = newm;
#pragma unroll
        for (int r = 0; r < 8; ++r) { O0[r] *= alpha; O1[r] *= alpha; }

        // rebuild P into B-fragment layout (32 keys x 16 queries)
        Frag Bp;
#pragma unroll
        for (int j = 0; j < 16; ++j) {
            const int src = mrow + (khalf ? 16 : 0);
            const float a0 = __shfl(P0[j & 7], src, 32);
            const float a1 = __shfl(P1[j & 7], src, 32);
            Bp.e[j] = (__bf16)((j & 8) ? a1 : a0);
        }

        // A_v (two 16-row d tiles): contiguous key runs of 8 from channel-major v
        Frag Av0, Av1;
        {
            const __bf16* p0 = vbase0 + k0;
            Av0.q[0] = *(const uint4*)p0;
            Av0.q[1] = *(const uint4*)(p0 + 16);
            const __bf16* p1 = vbase1 + k0;
            Av1.q[0] = *(const uint4*)p1;
            Av1.q[1] = *(const uint4*)(p1 + 16);
        }
        O0 = wmma_bf16(Av0, Bp, O0);
        O1 = wmma_bf16(Av1, Bp, O1);
    }

    const float inv = 1.0f / srun;
    Frag o;
#pragma unroll
    for (int r = 0; r < 8; ++r) o.e[r] = (__bf16)(O0[r] * inv);
    *(uint4*)(attnT + (size_t)(q0 + mrow) * CCH + hc + khalf * 8) = o.q[0];
#pragma unroll
    for (int r = 0; r < 8; ++r) o.e[r] = (__bf16)(O1[r] * inv);
    *(uint4*)(attnT + (size_t)(q0 + mrow) * CCH + hc + 16 + khalf * 8) = o.q[0];
}

// ---------- kernel 3: out = x + gamma * (Wo @ attn + bo) ----------
__global__ void k_proj(const __bf16* __restrict__ attnT,
                       const float* __restrict__ Wo, const float* __restrict__ bo,
                       const float* __restrict__ x, const float* __restrict__ gamma,
                       float* __restrict__ out) {
    const int m0 = (blockIdx.x & 15) * 16;
    const int n0 = (blockIdx.x >> 4) * 64;
    const int lane  = threadIdx.x & 31;
    const int mrow  = lane & 15;
    const int khalf = lane >> 4;

    f32x8 acc[4];
#pragma unroll
    for (int t = 0; t < 4; ++t)
#pragma unroll
        for (int r = 0; r < 8; ++r) acc[t][r] = 0.0f;

    for (int k0 = 0; k0 < CCH; k0 += 32) {
        Frag A;
        const float* wr = Wo + (size_t)(m0 + mrow) * CCH + k0 + khalf * 8;
#pragma unroll
        for (int j = 0; j < 8; ++j) A.e[j]     = (__bf16)wr[j];
#pragma unroll
        for (int j = 0; j < 8; ++j) A.e[8 + j] = (__bf16)wr[16 + j];
#pragma unroll
        for (int t = 0; t < 4; ++t) {
            Frag B;
            const uint4* s = (const uint4*)(attnT + (size_t)(n0 + t * 16 + mrow) * CCH
                                                  + k0 + khalf * 8);
            B.q[0] = s[0];
            B.q[1] = s[2];
            acc[t] = wmma_bf16(A, B, acc[t]);
        }
    }

    const float g = gamma[0];
#pragma unroll
    for (int t = 0; t < 4; ++t) {
        const int pos = n0 + t * 16 + mrow;
#pragma unroll
        for (int r = 0; r < 8; ++r) {
            const int chan = m0 + khalf * 8 + r;
            const size_t idx = (size_t)chan * NPOS + pos;
            out[idx] = x[idx] + g * (acc[t][r] + bo[chan]);
        }
    }
}

// ---------- host launch ----------
extern "C" void kernel_launch(void* const* d_in, const int* in_sizes, int n_in,
                              void* d_out, int out_size, void* d_ws, size_t ws_size,
                              hipStream_t stream) {
    const float* x     = (const float*)d_in[0];
    const float* Wq    = (const float*)d_in[1];
    const float* bq    = (const float*)d_in[2];
    const float* Wk    = (const float*)d_in[3];
    const float* bk    = (const float*)d_in[4];
    const float* Wv    = (const float*)d_in[5];
    const float* bv    = (const float*)d_in[6];
    const float* Wo    = (const float*)d_in[7];
    const float* bo    = (const float*)d_in[8];
    const float* gamma = (const float*)d_in[9];
    float* out = (float*)d_out;

    const size_t PLANE = (size_t)NPOS * CCH;      // 1M bf16 elements = 2 MB
    __bf16* xT    = (__bf16*)d_ws;
    __bf16* qT    = xT + PLANE;
    __bf16* kT    = qT + PLANE;
    __bf16* vC    = kT + PLANE;
    __bf16* attnT = vC + PLANE;                   // total 10 MB of d_ws

    // 0: transpose + bf16 convert
    k_xpose<<<dim3(NPOS / 32, CCH / 32), dim3(32, 8), 0, stream>>>(x, xT);
    // 1: QKV projections (y selects q/k/v)
    k_qkv<<<dim3((CCH / 16) * (NPOS / 64), 3), 32, 0, stream>>>(
        xT, Wq, Wk, Wv, bq, bk, bv, qT, kT, vC);
    // 2: flash attention, 8 heads x 256 query tiles
    k_attn<<<8 * (NPOS / 16), 32, 0, stream>>>(qT, kT, vC, attnT);
    // 3: output projection + residual
    k_proj<<<(CCH / 16) * (NPOS / 64), 32, 0, stream>>>(attnT, Wo, bo, x, gamma, out);
}
